// SimpleRNNNetwork_83631603187926
// MI455X (gfx1250) — matrix operand order repaired and checked by reading, not compiled, not run on hardware
//
#include <hip/hip_runtime.h>

typedef _Float16 half_t;
typedef __attribute__((ext_vector_type(16))) _Float16 v16h;
typedef __attribute__((ext_vector_type(8)))  _Float16 v8h;
typedef __attribute__((ext_vector_type(4)))  _Float16 v4h;
typedef __attribute__((ext_vector_type(2)))  _Float16 v2h;
typedef __attribute__((ext_vector_type(8)))  float    v8f;
typedef __attribute__((ext_vector_type(4)))  float    v4f;

#define F_IN    124
#define KPAD    128
#define H       64
#define T_STEPS 64
#define NBATCH  4096

// LDS row strides (in halves), padded against bank conflicts
#define WIH0_S 136   // [64][128]
#define WREC_S 72    // [64][64]
#define HBUF_S 72    // [16][64]
#define XBUF_S 136   // [16][128]

// ---- split workgroup barrier (CDNA5): signal early, wait late, put
// independent work inside the window.  Fallback keeps identical semantics.
#if __has_builtin(__builtin_amdgcn_s_barrier_signal) && \
    __has_builtin(__builtin_amdgcn_s_barrier_wait)
#define BAR_SIGNAL() do { \
    __builtin_amdgcn_fence(__ATOMIC_RELEASE, "workgroup"); \
    __builtin_amdgcn_s_barrier_signal(-1); } while (0)
#define BAR_WAIT() do { \
    __builtin_amdgcn_s_barrier_wait(-1); \
    __builtin_amdgcn_fence(__ATOMIC_ACQUIRE, "workgroup"); } while (0)
#else
#define BAR_SIGNAL() do { } while (0)
#define BAR_WAIT()   __syncthreads()
#endif

static __device__ __forceinline__ float fast_tanh(float x) {
#if __has_builtin(__builtin_amdgcn_tanhf)
    return __builtin_amdgcn_tanhf(x);          // v_tanh_f32 (gfx1250 trans op)
#else
    x = fminf(fmaxf(x, -15.0f), 15.0f);
    float e = __builtin_amdgcn_exp2f(x * 2.8853900817779268f);
    return (e - 1.0f) * __builtin_amdgcn_rcpf(e + 1.0f);
#endif
}

static __device__ __forceinline__ v16h combine16(v8h lo, v8h hi) {
    v16h r;
#pragma unroll
    for (int i = 0; i < 8; ++i) { r[i] = lo[i]; r[i + 8] = hi[i]; }
    return r;
}

// B-fragment (K x 16) for v_wmma_f32_16x16x32_f16 from LDS f16 matrix stored
// row-major [n][k]: lane holds column n = lane&15, halves i <-> k = g*16 + i.
static __device__ __forceinline__ v16h loadB(const half_t* W, int stride,
                                             int row, int c, int g) {
    const half_t* p = W + row * stride + c * 32 + g * 16;
    return combine16(*(const v8h*)p, *(const v8h*)(p + 8));
}

// A-fragment (16 x 32) from an LDS f16 tile [16][stride]: lane holds row
// m = lane&15; halves i <-> k = g*16 + i.
static __device__ __forceinline__ v16h loadA(const half_t* hb, int stride,
                                             int rowIdx, int c, int g) {
    const half_t* p = hb + rowIdx * stride + c * 32 + g * 16;
    return combine16(*(const v8h*)p, *(const v8h*)(p + 8));
}

static __device__ __forceinline__ v8f wmma(v16h a, v16h b, v8f c) {
    return __builtin_amdgcn_wmma_f32_16x16x32_f16(false, a, false, b,
                                                  (short)0, c, false, false);
}

// one 16x16 C-tile (f32, 8 vals/lane) -> bias + tanh -> f16 LDS h-buffer
// C layout: VGPR r holds row m = r + 8*g, col n = lane&15.
static __device__ __forceinline__ void store_htile(half_t* hb, v8f acc,
                                                   float bias, int col, int g) {
#pragma unroll
    for (int r = 0; r < 8; ++r)
        hb[(r + 8 * g) * HBUF_S + col] = (half_t)fast_tanh(acc[r] + bias);
}

// float4 -> packed 4xf16 8B store (v_cvt_pk_rtz_f16_f32 x2)
static __device__ __forceinline__ void store_x4(half_t* d, v4f v) {
    v2h lo = __builtin_bit_cast(v2h, __builtin_amdgcn_cvt_pkrtz(v[0], v[1]));
    v2h hi = __builtin_bit_cast(v2h, __builtin_amdgcn_cvt_pkrtz(v[2], v[3]));
    v4h o; o[0] = lo[0]; o[1] = lo[1]; o[2] = hi[0]; o[3] = hi[1];
    *(v4h*)d = o;
}

__global__ __launch_bounds__(128, 1)
void rnn2_fused_kernel(const float* __restrict__ x,
                       const float* __restrict__ Wih0, const float* __restrict__ Whh0,
                       const float* __restrict__ bih0, const float* __restrict__ bhh0,
                       const float* __restrict__ Wih1, const float* __restrict__ Whh1,
                       const float* __restrict__ bih1, const float* __restrict__ bhh1,
                       const float* __restrict__ Wfc,  const float* __restrict__ bfc,
                       float* __restrict__ out) {
    __shared__ half_t sWih0[H * WIH0_S];
    __shared__ half_t sWhh0[H * WREC_S];
    __shared__ half_t sWih1[H * WREC_S];
    __shared__ half_t sWhh1[H * WREC_S];
    __shared__ float  sBias0[H], sBias1[H], sWfc[H];
    __shared__ half_t sH0[16 * HBUF_S];
    __shared__ half_t sH1[16 * HBUF_S];
    __shared__ half_t sXbuf[2][16 * XBUF_S];

    const int tid  = threadIdx.x;       // 0..127 (4 waves)
    const int nt   = tid >> 5;          // wave id == N-tile this wave owns
    const int lane = tid & 31;
    const int nidx = lane & 15;
    const int g    = (lane >> 4) & 1;

    // x staging mapping: 16 rows x 31 float4 per timestep; 8 threads/row
    const int srow = tid >> 3;          // 0..15
    const int sq   = tid & 7;           // base float4 slot
    const float* xsrc =
        x + (size_t)(blockIdx.x * 16 + srow) * T_STEPS * F_IN;

    // ---- stage weights as f16 into LDS (one-time, whole block) ----
    for (int i = tid; i < H * KPAD; i += 128) {
        int r = i >> 7, c = i & 127;
        sWih0[r * WIH0_S + c] =
            (c < F_IN) ? (half_t)Wih0[r * F_IN + c] : (half_t)0.0f;
    }
    for (int i = tid; i < H * H; i += 128) {
        int r = i >> 6, c = i & 63;
        sWhh0[r * WREC_S + c] = (half_t)Whh0[i];
        sWih1[r * WREC_S + c] = (half_t)Wih1[i];
        sWhh1[r * WREC_S + c] = (half_t)Whh1[i];
    }
    for (int i = tid; i < H; i += 128) {
        sBias0[i] = bih0[i] + bhh0[i];
        sBias1[i] = bih1[i] + bhh1[i];
        sWfc[i]   = Wfc[i];
    }
    // zero the K-pad columns of both x buffers (staging never touches them)
    for (int i = tid; i < 2 * 16; i += 128) {
        int b = i >> 4, r = i & 15;
#pragma unroll
        for (int c = F_IN; c < KPAD; ++c) sXbuf[b][r * XBUF_S + c] = (half_t)0.0f;
    }

    // stage t=0 into buffer 0; preload t=1 into registers (pfA)
    v4f pfA[4] = {}, pfB[4] = {};
#pragma unroll
    for (int j = 0; j < 4; ++j) {
        int q = sq + 8 * j;
        if (q < 31) {
            v4f v = *(const v4f*)(xsrc + 4 * q);
            store_x4(&sXbuf[0][srow * XBUF_S + 4 * q], v);
            pfA[j] = *(const v4f*)(xsrc + F_IN + 4 * q);
        }
    }
    __syncthreads();

    // ---- hoist this wave's weight B-fragments into registers (loop-invariant)
    const int brow = nt * 16 + nidx;
    v16h bWih0[4], bWhh0[2], bWih1[2], bWhh1[2];
#pragma unroll
    for (int c = 0; c < 4; ++c) bWih0[c] = loadB(sWih0, WIH0_S, brow, c, g);
#pragma unroll
    for (int c = 0; c < 2; ++c) {
        bWhh0[c] = loadB(sWhh0, WREC_S, brow, c, g);
        bWih1[c] = loadB(sWih1, WREC_S, brow, c, g);
        bWhh1[c] = loadB(sWhh1, WREC_S, brow, c, g);
    }
    const float bias0 = sBias0[brow];
    const float bias1 = sBias1[brow];

    v16h h0a[2] = {};   // h state A-fragments (zero initial state)
    v16h h1a[2] = {};

    // one timestep; pfD holds x_{t+1} (drained here), pfL receives x_{t+2}
    auto step = [&](int t, int cur, v4f (&pfD)[4], v4f (&pfL)[4]) {
        // issue distance-2 global loads first: in flight for ~2 full steps
        if (t + 2 < T_STEPS) {
            const float* src = xsrc + (t + 2) * F_IN;
#pragma unroll
            for (int j = 0; j < 4; ++j) {
                int q = sq + 8 * j;
                if (q < 31) pfL[j] = *(const v4f*)(src + 4 * q);
            }
        }

        // ---- layer 0 ----
        v16h xa[4];
#pragma unroll
        for (int c = 0; c < 4; ++c) xa[c] = loadA(sXbuf[cur], XBUF_S, nidx, c, g);
        v8f acc0 = {};
        acc0 = wmma(h0a[0], bWhh0[0], acc0);
        acc0 = wmma(h0a[1], bWhh0[1], acc0);
#pragma unroll
        for (int c = 0; c < 4; ++c) acc0 = wmma(xa[c], bWih0[c], acc0);
        store_htile(sH0, acc0, bias0, brow, g);

        BAR_SIGNAL();
        // --- independent work inside the barrier window ---
        v8f acc1 = {};
        acc1 = wmma(h1a[0], bWhh1[0], acc1);       // register-resident operands
        acc1 = wmma(h1a[1], bWhh1[1], acc1);
        if (t + 1 < T_STEPS) {                      // drain x_{t+1}
#pragma unroll
            for (int j = 0; j < 4; ++j) {
                int q = sq + 8 * j;
                if (q < 31)
                    store_x4(&sXbuf[cur ^ 1][srow * XBUF_S + 4 * q], pfD[j]);
            }
        }
        BAR_WAIT();

        // ---- layer 1 ----
        h0a[0] = loadA(sH0, HBUF_S, nidx, 0, g);
        h0a[1] = loadA(sH0, HBUF_S, nidx, 1, g);
        acc1 = wmma(h0a[0], bWih1[0], acc1);
        acc1 = wmma(h0a[1], bWih1[1], acc1);
        store_htile(sH1, acc1, bias1, brow, g);

        BAR_SIGNAL();
        BAR_WAIT();
        h1a[0] = loadA(sH1, HBUF_S, nidx, 0, g);
        h1a[1] = loadA(sH1, HBUF_S, nidx, 1, g);
    };

    for (int t2 = 0; t2 < T_STEPS; t2 += 2) {
        step(t2,     0, pfA, pfB);
        step(t2 + 1, 1, pfB, pfA);
    }

    // ---- head: out[m] = h1[m,:] . Wfc + bfc  (final h1 is in sH1) ----
    __syncthreads();
    if (tid < 16) {
        float s = bfc[0];
#pragma unroll 8
        for (int k = 0; k < H; ++k)
            s += (float)sH1[tid * HBUF_S + k] * sWfc[k];
        out[blockIdx.x * 16 + tid] = s;
    }
}

extern "C" void kernel_launch(void* const* d_in, const int* in_sizes, int n_in,
                              void* d_out, int out_size, void* d_ws, size_t ws_size,
                              hipStream_t stream) {
    const float* x    = (const float*)d_in[0];
    const float* Wih0 = (const float*)d_in[1];
    const float* Whh0 = (const float*)d_in[2];
    const float* bih0 = (const float*)d_in[3];
    const float* bhh0 = (const float*)d_in[4];
    const float* Wih1 = (const float*)d_in[5];
    const float* Whh1 = (const float*)d_in[6];
    const float* bih1 = (const float*)d_in[7];
    const float* bhh1 = (const float*)d_in[8];
    const float* Wfc  = (const float*)d_in[9];
    const float* bfc  = (const float*)d_in[10];
    float* out = (float*)d_out;

    dim3 grid(NBATCH / 16);   // 256 blocks x 4 waves = 1024 waves
    dim3 block(128);          // wave w owns hidden columns [16w, 16w+16)
    rnn2_fused_kernel<<<grid, block, 0, stream>>>(x, Wih0, Whh0, bih0, bhh0,
                                                  Wih1, Whh1, bih1, bhh1,
                                                  Wfc, bfc, out);
}
